// SimpleCrossAttention_73289321939271
// MI455X (gfx1250) — matrix operand order repaired and checked by reading, not compiled, not run on hardware
//
#include <hip/hip_runtime.h>
#include <hip/hip_bf16.h>

#define B_   2
#define C_   256
#define L_   4096
#define H_   8
#define D_   32
#define EPS_ 1e-5f

typedef __attribute__((ext_vector_type(16))) __bf16 v16bf;
typedef __attribute__((ext_vector_type(8)))  float  v8f;

union BF16Frag { unsigned short u[16]; v16bf v; };

__device__ __forceinline__ unsigned short f2bf(float f) {
    unsigned u = __builtin_bit_cast(unsigned, f);
    u += 0x7FFFu + ((u >> 16) & 1u);           // round-to-nearest-even
    return (unsigned short)(u >> 16);
}

// A-fragment (16x32 bf16). M = lane&15. VGPR v holds K = 16*hi+2v , K+1.
__device__ __forceinline__ v16bf load_a_frag(const unsigned short* base, int rs, int lane) {
    BF16Frag f;
    const int m = lane & 15, hi = lane >> 4;
#pragma unroll
    for (int v = 0; v < 8; ++v) {
        const int k = hi * 16 + 2 * v;
        f.u[2 * v]     = base[m * rs + k];
        f.u[2 * v + 1] = base[m * rs + k + 1];
    }
    return f.v;
}

// B-fragment (32x16 bf16). N = lane&15; lanes 0-15 hold K=0..15, lanes 16-31 K=16..31.
// Transposed storage: element (k,n) at base[n*rs + k]  (pair k,k+1 contiguous).
__device__ __forceinline__ v16bf load_b_frag(const unsigned short* base, int rs, int lane) {
    BF16Frag f;
    const int n = lane & 15, hi = lane >> 4;
#pragma unroll
    for (int v = 0; v < 8; ++v) {
        const int k = hi * 16 + 2 * v;
        f.u[2 * v]     = base[n * rs + k];
        f.u[2 * v + 1] = base[n * rs + k + 1];
    }
    return f.v;
}

// B-fragment from row-major [K][N] storage: element (k,n) at base[k*rs + n].
// Used for tiles staged by async direct-to-LDS copies (no transpose in flight).
__device__ __forceinline__ v16bf load_b_frag_rm(const unsigned short* base, int rs, int lane) {
    BF16Frag f;
    const int n = lane & 15, hi = lane >> 4;
#pragma unroll
    for (int v = 0; v < 8; ++v) {
        const int k = hi * 16 + 2 * v;
        f.u[2 * v]     = base[k * rs + n];
        f.u[2 * v + 1] = base[(k + 1) * rs + n];
    }
    return f.v;
}

__device__ __forceinline__ v8f bf16_wmma(v16bf a, v16bf b, v8f c) {
    return __builtin_amdgcn_wmma_f32_16x16x32_bf16(false, a, false, b, (short)0, c,
                                                   false, false);
}

// ---------------------------------------------------------------------------
// Kernel 1: QKV = W_qkv * X + b  (per batch), scatter to Q^T / K / V^T in bf16
// ---------------------------------------------------------------------------
__global__ __launch_bounds__(256)
void qkv_gemm_kernel(const float* __restrict__ x, const float* __restrict__ w_qkv,
                     const float* __restrict__ b_qkv,
                     unsigned short* __restrict__ QT,   // [B][H][L][D]  (q * scale)
                     unsigned short* __restrict__ Kg,   // [B][H][D][L]
                     unsigned short* __restrict__ VTg)  // [B][H][L][D]
{
    __shared__ unsigned short As[64][34];    // W tile  [o'][c']
    __shared__ unsigned short BsT[128][34];  // X tile  [p'][c'] (transposed)

    const int tid  = threadIdx.x;
    const int lane = tid & 31, wave = tid >> 5;
    const int wx = wave & 3, wy = wave >> 2;        // 2x4 wave grid -> 32x32 per wave
    const int pbase = blockIdx.x * 128;
    const int obase = blockIdx.y * 64;
    const int b     = blockIdx.z;

    v8f zero = {};
    v8f acc[2][2];
    acc[0][0] = zero; acc[0][1] = zero; acc[1][0] = zero; acc[1][1] = zero;

    for (int cbase = 0; cbase < C_; cbase += 32) {
        __syncthreads();
#pragma unroll
        for (int rep = 0; rep < 8; ++rep) {          // stage W 64x32
            int idx = rep * 256 + tid;
            int o = idx >> 5, c = idx & 31;
            As[o][c] = f2bf(w_qkv[(size_t)(obase + o) * C_ + cbase + c]);
        }
#pragma unroll
        for (int rep = 0; rep < 16; ++rep) {         // stage X 32x128 (transposed)
            int idx = rep * 256 + tid;
            int c = idx >> 7, p = idx & 127;
            BsT[p][c] = f2bf(x[((size_t)b * C_ + cbase + c) * L_ + pbase + p]);
        }
        __syncthreads();

        v16bf a0 = load_a_frag(&As[wy * 32][0], 34, lane);
        v16bf a1 = load_a_frag(&As[wy * 32 + 16][0], 34, lane);
        v16bf b0 = load_b_frag(&BsT[wx * 32][0], 34, lane);
        v16bf b1 = load_b_frag(&BsT[wx * 32 + 16][0], 34, lane);
        acc[0][0] = bf16_wmma(a0, b0, acc[0][0]);
        acc[0][1] = bf16_wmma(a0, b1, acc[0][1]);
        acc[1][0] = bf16_wmma(a1, b0, acc[1][0]);
        acc[1][1] = bf16_wmma(a1, b1, acc[1][1]);
    }

    const float qscale = 0.1767766952966369f;        // 1/sqrt(32)
    const int hi = lane >> 4, n = lane & 15;
#pragma unroll
    for (int mi = 0; mi < 2; ++mi)
#pragma unroll
        for (int ni = 0; ni < 2; ++ni)
#pragma unroll
            for (int r = 0; r < 8; ++r) {
                int o = obase + wy * 32 + mi * 16 + r + 8 * hi;
                int p = pbase + wx * 32 + ni * 16 + n;
                float val = acc[mi][ni][r] + b_qkv[o];
                if (o < C_) {                               // Q -> Q^T, pre-scaled
                    int h = o >> 5, dd = o & 31;
                    QT[(((size_t)b * H_ + h) * L_ + p) * D_ + dd] = f2bf(val * qscale);
                } else if (o < 2 * C_) {                    // K -> d-major
                    int oo = o - C_; int h = oo >> 5, dd = oo & 31;
                    Kg[(((size_t)b * H_ + h) * D_ + dd) * L_ + p] = f2bf(val);
                } else {                                    // V -> V^T
                    int oo = o - 2 * C_; int h = oo >> 5, dd = oo & 31;
                    VTg[(((size_t)b * H_ + h) * L_ + p) * D_ + dd] = f2bf(val);
                }
            }
}

// ---------------------------------------------------------------------------
// Kernel 2: flash attention. One WG per (b,h, 128 query rows); 16 rows/wave.
// K/V chunks staged with GLOBAL_LOAD_ASYNC_TO_LDS_B64 (ASYNCcnt), no VGPR
// round-trip; producer/consumer handshake = s_wait_asynccnt 0 + barrier.
// ---------------------------------------------------------------------------
__global__ __launch_bounds__(256)
void flash_attn_kernel(const unsigned short* __restrict__ QT,
                       const unsigned short* __restrict__ Kg,
                       const unsigned short* __restrict__ VTg,
                       unsigned short* __restrict__ AO)   // [B][C][L]
{
    __shared__ __align__(16) unsigned short Ks[32][36];   // [d][j']  (global order)
    __shared__ __align__(16) unsigned short Vs[32][36];   // [j'][d]  (global order)
    __shared__ unsigned short Ps[8][16][34];              // per-wave P tile [i'][j']

    const int tid = threadIdx.x, lane = tid & 31, wave = tid >> 5;
    const int bh = blockIdx.y;                  // b*H + h
    const int qbase = blockIdx.x * 128 + wave * 16;

    const unsigned short* Qp = QT + ((size_t)bh * L_ + qbase) * D_;
    const unsigned short* Kp = Kg + (size_t)bh * D_ * L_;
    const unsigned short* Vp = VTg + (size_t)bh * L_ * D_;

    const int hi = lane >> 4, col = lane & 15;

    // Q fragment: row-contiguous in d, loaded once.
    BF16Frag qf;
    {
        const int m = lane & 15;
#pragma unroll
        for (int v = 0; v < 8; ++v) {
            const int k = hi * 16 + 2 * v;
            qf.u[2 * v]     = Qp[m * D_ + k];
            qf.u[2 * v + 1] = Qp[m * D_ + k + 1];
        }
    }

    v8f zero = {};
    v8f o0 = zero, o1 = zero;
    float mrow[8], lrow[8];
#pragma unroll
    for (int r = 0; r < 8; ++r) { mrow[r] = -3.0e38f; lrow[r] = 0.0f; }

    unsigned short* Pw = &Ps[wave][0][0];

    // per-thread async-copy geometry: 256 threads x 8 bytes = one 2 KB chunk
    const int crow = tid >> 3;                  // 0..31
    const int cq   = (tid & 7) * 4;             // quad of bf16 within the row
    const unsigned kldsoff = (unsigned)(size_t)&Ks[crow][cq];
    const unsigned vldsoff = (unsigned)(size_t)&Vs[crow][cq];

    for (int jb = 0; jb < L_; jb += 32) {
        __syncthreads();                        // previous chunk fully consumed
        {
            unsigned kgo = (unsigned)(((size_t)crow * L_ + jb + cq) * 2);
            asm volatile("global_load_async_to_lds_b64 %0, %1, %2"
                         :: "v"(kldsoff), "v"(kgo), "s"(Kp) : "memory");
            unsigned vgo = (unsigned)(((size_t)(jb + crow) * D_ + cq) * 2);
            asm volatile("global_load_async_to_lds_b64 %0, %1, %2"
                         :: "v"(vldsoff), "v"(vgo), "s"(Vp) : "memory");
        }
        if (jb + 32 < L_) {                     // global_prefetch_b8 of next chunk
            __builtin_prefetch(Kp + (size_t)crow * L_ + jb + 32 + cq, 0, 1);
            __builtin_prefetch(Vp + (size_t)(jb + 32 + crow) * D_ + cq, 0, 1);
        }
        asm volatile("s_wait_asynccnt 0x0" ::: "memory");
        __syncthreads();                        // chunk visible to all waves

        // S = (Q^T)(K) for 16 rows x 32 cols : two WMMAs
        v16bf kb0 = load_b_frag_rm(&Ks[0][0], 36, lane);
        v16bf kb1 = load_b_frag_rm(&Ks[0][16], 36, lane);
        v8f s0 = bf16_wmma(qf.v, kb0, zero);
        v8f s1 = bf16_wmma(qf.v, kb1, zero);

        // online softmax update (row stats replicated across each 16-lane group)
#pragma unroll
        for (int r = 0; r < 8; ++r) {
            float a = s0[r], c = s1[r];
            float mx = fmaxf(a, c);
#pragma unroll
            for (int off = 8; off >= 1; off >>= 1)
                mx = fmaxf(mx, __shfl_xor(mx, off, 32));
            float mnew  = fmaxf(mrow[r], mx);
            float alpha = __expf(mrow[r] - mnew);
            float e0 = __expf(a - mnew);
            float e1 = __expf(c - mnew);
            float rs = e0 + e1;
#pragma unroll
            for (int off = 8; off >= 1; off >>= 1)
                rs += __shfl_xor(rs, off, 32);
            lrow[r] = lrow[r] * alpha + rs;
            mrow[r] = mnew;
            o0[r] *= alpha;
            o1[r] *= alpha;
            Pw[(r + 8 * hi) * 34 + col]      = f2bf(e0);
            Pw[(r + 8 * hi) * 34 + 16 + col] = f2bf(e1);
        }

        // O += P * V^T  (two 16-wide d tiles)
        v16bf pf  = load_a_frag(Pw, 34, lane);
        v16bf vb0 = load_b_frag_rm(&Vs[0][0], 36, lane);
        v16bf vb1 = load_b_frag_rm(&Vs[0][16], 36, lane);
        o0 = bf16_wmma(pf, vb0, o0);
        o1 = bf16_wmma(pf, vb1, o1);
    }

    // epilogue: normalize and scatter to AO[b][c][p] (c-major for the proj GEMM)
    const int b = bh >> 3, h = bh & 7;
#pragma unroll
    for (int r = 0; r < 8; ++r) {
        float inv = 1.0f / lrow[r];
        int i = qbase + r + 8 * hi;
        AO[((size_t)b * C_ + h * D_ + col) * L_ + i]      = f2bf(o0[r] * inv);
        AO[((size_t)b * C_ + h * D_ + 16 + col) * L_ + i] = f2bf(o1[r] * inv);
    }
}

// ---------------------------------------------------------------------------
// Kernel 3: out = BN(W_proj * AO + b_proj)
// ---------------------------------------------------------------------------
__global__ __launch_bounds__(256)
void proj_bn_kernel(const unsigned short* __restrict__ AO,
                    const float* __restrict__ w_proj, const float* __restrict__ b_proj,
                    const float* __restrict__ gamma, const float* __restrict__ beta,
                    const float* __restrict__ rmean, const float* __restrict__ rvar,
                    float* __restrict__ out)
{
    __shared__ unsigned short As[64][34];
    __shared__ unsigned short BsT[128][34];

    const int tid  = threadIdx.x;
    const int lane = tid & 31, wave = tid >> 5;
    const int wx = wave & 3, wy = wave >> 2;
    const int pbase = blockIdx.x * 128;
    const int obase = blockIdx.y * 64;
    const int b     = blockIdx.z;

    v8f zero = {};
    v8f acc[2][2];
    acc[0][0] = zero; acc[0][1] = zero; acc[1][0] = zero; acc[1][1] = zero;

    for (int cbase = 0; cbase < C_; cbase += 32) {
        __syncthreads();
#pragma unroll
        for (int rep = 0; rep < 8; ++rep) {
            int idx = rep * 256 + tid;
            int o = idx >> 5, c = idx & 31;
            As[o][c] = f2bf(w_proj[(size_t)(obase + o) * C_ + cbase + c]);
        }
#pragma unroll
        for (int rep = 0; rep < 16; ++rep) {
            int idx = rep * 256 + tid;
            int c = idx >> 7, p = idx & 127;
            BsT[p][c] = AO[((size_t)b * C_ + cbase + c) * L_ + pbase + p];
        }
        __syncthreads();

        v16bf a0 = load_a_frag(&As[wy * 32][0], 34, lane);
        v16bf a1 = load_a_frag(&As[wy * 32 + 16][0], 34, lane);
        v16bf b0 = load_b_frag(&BsT[wx * 32][0], 34, lane);
        v16bf b1 = load_b_frag(&BsT[wx * 32 + 16][0], 34, lane);
        acc[0][0] = bf16_wmma(a0, b0, acc[0][0]);
        acc[0][1] = bf16_wmma(a0, b1, acc[0][1]);
        acc[1][0] = bf16_wmma(a1, b0, acc[1][0]);
        acc[1][1] = bf16_wmma(a1, b1, acc[1][1]);
    }

    const int hi = lane >> 4, n = lane & 15;
#pragma unroll
    for (int mi = 0; mi < 2; ++mi)
#pragma unroll
        for (int ni = 0; ni < 2; ++ni)
#pragma unroll
            for (int r = 0; r < 8; ++r) {
                int o = obase + wy * 32 + mi * 16 + r + 8 * hi;
                int p = pbase + wx * 32 + ni * 16 + n;
                float val = acc[mi][ni][r] + b_proj[o];
                val = (val - rmean[o]) * (gamma[o] * rsqrtf(rvar[o] + EPS_)) + beta[o];
                out[((size_t)b * C_ + o) * L_ + p] = val;
            }
}

// ---------------------------------------------------------------------------
extern "C" void kernel_launch(void* const* d_in, const int* in_sizes, int n_in,
                              void* d_out, int out_size, void* d_ws, size_t ws_size,
                              hipStream_t stream) {
    const float* x      = (const float*)d_in[0];
    const float* w_qkv  = (const float*)d_in[1];
    const float* b_qkv  = (const float*)d_in[2];
    const float* w_proj = (const float*)d_in[3];
    const float* b_proj = (const float*)d_in[4];
    const float* gamma  = (const float*)d_in[5];
    const float* beta   = (const float*)d_in[6];
    const float* rmean  = (const float*)d_in[7];
    const float* rvar   = (const float*)d_in[8];
    float* out = (float*)d_out;

    const size_t nQ = (size_t)B_ * H_ * L_ * D_;       // 2M bf16 elements each
    unsigned short* QT  = (unsigned short*)d_ws;
    unsigned short* Kg  = QT + nQ;
    unsigned short* VTg = Kg + nQ;
    unsigned short* AO  = VTg + nQ;                    // total 16 MB of d_ws

    qkv_gemm_kernel<<<dim3(L_ / 128, (3 * C_) / 64, B_), 256, 0, stream>>>(
        x, w_qkv, b_qkv, QT, Kg, VTg);
    flash_attn_kernel<<<dim3(L_ / 128, B_ * H_), 256, 0, stream>>>(QT, Kg, VTg, AO);
    proj_bn_kernel<<<dim3(L_ / 128, C_ / 64, B_), 256, 0, stream>>>(
        AO, w_proj, b_proj, gamma, beta, rmean, rvar, out);
}
